// Decoder_24026047053980
// MI455X (gfx1250) — matrix operand order repaired and checked by reading, not compiled, not run on hardware
//
#include <hip/hip_runtime.h>

// Decoder: GRU + per-head single-query attention + MLP head, T=288 sequential
// steps, batch-parallel (2048). Strategy for MI455X:
//  1) prep kernel: convert encoder values + precomputed keys to bf16 (151 MB
//     total -> fits 192 MB L2, so the 288 per-step re-reads are L2-resident
//     instead of 87 GB of HBM traffic).
//  2) step kernel: 128 WGs x 256 thr, 16-batch tile per WG, whole 288-step
//     recurrence in one launch (batch elements are independent). GRU gates,
//     attention scores (block-diagonal Q trick) and MLP use
//     v_wmma_f32_16x16x32_bf16; softmax/ctx on VALU with packed bf16 dwords,
//     bit-shift bf16->f32 conversion, and v_rcp_f32 instead of IEEE division.
// Workspace: needs 2 * 2048*288*64 * 2B = 151 MB of d_ws.

#define BSZ  2048
#define SEQ  288
#define TT   288
#define HD   64
#define NH   8
#define DSZ  29
#define MT   16   // batch tile (WMMA M)

typedef __attribute__((ext_vector_type(16))) __bf16 v16bf;
typedef __attribute__((ext_vector_type(8)))  float  v8f;

union FragAB { v16bf v; uint4 q[2]; };

// bf16 <-> f32 by bit manipulation (1 VALU op instead of v_cvt chains)
__device__ __forceinline__ float bflo(unsigned int p) {
  return __uint_as_float(p << 16);
}
__device__ __forceinline__ float bfhi(unsigned int p) {
  return __uint_as_float(p & 0xffff0000u);
}
__device__ __forceinline__ unsigned int f2bfbits(float x) {  // round-to-nearest-even
  unsigned int u = __float_as_uint(x);
  u += 0x7fffu + ((u >> 16) & 1u);
  return u >> 16;
}

__device__ __forceinline__ float fast_rcp(float x) {
  return __builtin_amdgcn_rcpf(x);
}
__device__ __forceinline__ float fast_sigmoid(float x) {
  return fast_rcp(1.f + __expf(-x));
}
__device__ __forceinline__ float fast_tanh(float x) {
  float e = __expf(-2.f * fabsf(x));        // in (0,1], no overflow
  float t = (1.f - e) * fast_rcp(1.f + e);
  return copysignf(t, x);
}

// A-fragment (16x32 bf16) from a row-major bf16 row: lane L
// (row = L&15, hi = L>>4) holds K = hi*16..hi*16+15 (matches B fragments),
// i.e. two contiguous 16B chunks -> two b128 loads.
__device__ __forceinline__ v16bf ld_frag16(const __bf16* rowbase, int hi) {
  FragAB f;
  f.q[0] = *(const uint4*)(const void*)(rowbase + hi * 16);
  f.q[1] = *(const uint4*)(const void*)(rowbase + hi * 16 + 8);
  return f.v;
}

__device__ __forceinline__ v8f wmma_bf16(v16bf a, v16bf b, v8f c) {
  return __builtin_amdgcn_wmma_f32_16x16x32_bf16(false, a, false, b, (short)0,
                                                 c, false, false);
}

// B-fragment (32x16 bf16): lane L holds column n = L&15; element j holds
// K = K0 + hi*16 + j (contiguous in j -> contiguous weight reads).
__device__ __forceinline__ v16bf build_bfrag(const float* wrow, int K0,
                                             int Kmax, int hi) {
  FragAB f;
#pragma unroll
  for (int j = 0; j < 16; ++j) {
    int K = K0 + hi * 16 + j;
    f.v[j] = (__bf16)((K < Kmax) ? wrow[K] : 0.0f);
  }
  return f.v;
}

struct __align__(16) Smem {
  __bf16 scores[MT][NH][SEQ];   // 73728 B (holds unnormalized exp after softmax)
  float  g_rz[MT][2 * HD];      // fused gx+gh for r,z
  float  g_nx[MT][HD];          // gx for n
  float  g_nh[MT][HD];          // gh for n
  float  hcur[MT][HD];
  float  h1[MT][32];
  float  outv[MT];
  float  brz[2 * HD];
  float  bihn[HD];
  float  bhhn[HD];
  float  b1s[32];
  float  w2s[32];
  __bf16 hb[MT][HD];
  __bf16 xb[MT][32];            // 29 features zero-padded to 32
  __bf16 ctxb[MT][HD];
};                              // ~100.3 KB -> 3 WGs / WGP (320 KB LDS)

// ---------------------------------------------------------------------------
// Kernel 1: bf16 val64[b][s][64] and key64[b][s][64] = val @ Wk^T + bk
// ---------------------------------------------------------------------------
__global__ __launch_bounds__(256) void prep_kernel(const float* __restrict__ enc,
                                                   const float* __restrict__ Wk,
                                                   const float* __restrict__ bk,
                                                   __bf16* __restrict__ valb,
                                                   __bf16* __restrict__ keyb) {
  int tid = threadIdx.x;
  int k = tid & 63;
  long row = (long)blockIdx.x * 4 + (tid >> 6);  // row = b*SEQ + s
  if (row >= (long)BSZ * SEQ) return;
  int b = (int)(row / SEQ);
  int s = (int)(row % SEQ);
  const float* e = enc + ((size_t)s * BSZ + b) * HD;  // enc is (SEQ,BS,H)
  float v = e[k];
  int h = k >> 3, d = k & 7;
  float acc = bk[d];
#pragma unroll
  for (int dp = 0; dp < 8; ++dp) acc += e[h * 8 + dp] * Wk[d * 8 + dp];
  valb[row * HD + k] = (__bf16)v;
  keyb[row * HD + k] = (__bf16)acc;
}

// ---------------------------------------------------------------------------
// Kernel 2: full 288-step recurrence for a 16-element batch tile
// ---------------------------------------------------------------------------
__global__ __launch_bounds__(256, 1) void decoder_step_kernel(
    const float* __restrict__ x_dec, const float* __restrict__ hidden,
    const float* __restrict__ W_ih, const float* __restrict__ W_hh,
    const float* __restrict__ b_ih, const float* __restrict__ b_hh,
    const float* __restrict__ W1, const float* __restrict__ b1,
    const float* __restrict__ W2, const float* __restrict__ b2,
    const __bf16* __restrict__ keyb, const __bf16* __restrict__ valb,
    float* __restrict__ out) {
  __shared__ Smem sm;
  const int tid = threadIdx.x;
  const int wave = tid >> 5;
  const int lane = tid & 31;
  const int ln16 = lane & 15;
  const int hi = lane >> 4;
  const int bbase = blockIdx.x * MT;

  // ---- one-time staging ----
  if (tid < 128) {
    sm.brz[tid] = b_ih[tid] + b_hh[tid];
  } else if (tid < 192) {
    sm.bihn[tid - 128] = b_ih[tid];
    sm.bhhn[tid - 128] = b_hh[tid];
  } else if (tid < 224) {
    int j = tid - 192;
    sm.b1s[j] = b1[j];
    sm.w2s[j] = W2[j];
  }
  const float b2v = b2[0];
  for (int i = tid; i < MT * HD; i += 256) {
    int b = i >> 6, k = i & 63;
    float hv = hidden[(size_t)(bbase + b) * HD + k];
    sm.hcur[b][k] = hv;
    sm.hb[b][k] = (__bf16)hv;
  }
  for (int i = tid; i < MT * 32; i += 256) {  // step 0 uses x_dec[:,0,:] fully
    int b = i >> 5, f = i & 31;
    float xv = (f < DSZ) ? x_dec[(size_t)(bbase + b) * (TT + 1) * DSZ + f] : 0.f;
    sm.xb[b][f] = (__bf16)xv;
  }

  // ---- persistent weight B-fragments (held in VGPRs across all steps) ----
  // gates = x@W_ih^T + h@W_hh^T : N=192 = 12 tiles of 16.
  // Tile 'wave' (0..7) covers r/z cols [0,128); tile 'wave+8' (waves 0-3)
  // covers n cols [128,192) and needs split gx/gh accumulators.
  v16bf bwih0, bwhh0a, bwhh0b;
  v16bf bwih1 = {}, bwhh1a = {}, bwhh1b = {};
  {
    int col = wave * 16 + ln16;
    bwih0 = build_bfrag(W_ih + (size_t)col * DSZ, 0, DSZ, hi);
    bwhh0a = build_bfrag(W_hh + (size_t)col * HD, 0, HD, hi);
    bwhh0b = build_bfrag(W_hh + (size_t)col * HD, 32, HD, hi);
  }
  if (wave < 4) {
    int col = (wave + 8) * 16 + ln16;
    bwih1 = build_bfrag(W_ih + (size_t)col * DSZ, 0, DSZ, hi);
    bwhh1a = build_bfrag(W_hh + (size_t)col * HD, 0, HD, hi);
    bwhh1b = build_bfrag(W_hh + (size_t)col * HD, 32, HD, hi);
  }
  v16bf bw1a = {}, bw1b = {};
  if (wave < 2) {  // MLP layer-1: N=32 = 2 tiles, waves 0/1
    int col = wave * 16 + ln16;
    bw1a = build_bfrag(W1 + (size_t)col * HD, 0, HD, hi);
    bw1b = build_bfrag(W1 + (size_t)col * HD, 32, HD, hi);
  }
  __syncthreads();

  for (int t = 0; t < TT; ++t) {
    // ---- Phase A: GRU gate GEMMs (WMMA) ----
    {
      v16bf ax = ld_frag16(&sm.xb[ln16][0], hi);    // 16x32, K=29 padded
      v16bf ah0 = ld_frag16(&sm.hb[ln16][0], hi);   // K chunk 0
      v16bf ah1 = ld_frag16(&sm.hb[ln16][32], hi);  // K chunk 1
      {
        v8f c = {};
        c = wmma_bf16(ax, bwih0, c);    // gx
        c = wmma_bf16(ah0, bwhh0a, c);  // + gh (fused: r,z use gx+gh)
        c = wmma_bf16(ah1, bwhh0b, c);
        int col = wave * 16 + ln16;
#pragma unroll
        for (int v = 0; v < 8; ++v) sm.g_rz[hi * 8 + v][col] = c[v];
      }
      if (wave < 4) {  // n-gate tile: keep gx and gh separate
        v8f cx = {};
        cx = wmma_bf16(ax, bwih1, cx);
        v8f ch = {};
        ch = wmma_bf16(ah0, bwhh1a, ch);
        ch = wmma_bf16(ah1, bwhh1b, ch);
        int col = wave * 16 + ln16;  // (wave+8)*16 - 128
#pragma unroll
        for (int v = 0; v < 8; ++v) {
          sm.g_nx[hi * 8 + v][col] = cx[v];
          sm.g_nh[hi * 8 + v][col] = ch[v];
        }
      }
    }
    __syncthreads();

    // ---- Phase B: gate nonlinearity + h update (fast sigmoid/tanh) ----
    for (int i = tid; i < MT * HD; i += 256) {
      int b = i >> 6, j = i & 63;
      float r = fast_sigmoid(sm.g_rz[b][j] + sm.brz[j]);
      float z = fast_sigmoid(sm.g_rz[b][64 + j] + sm.brz[64 + j]);
      float n = fast_tanh(sm.g_nx[b][j] + sm.bihn[j] +
                          r * (sm.g_nh[b][j] + sm.bhhn[j]));
      float hn = (1.f - z) * n + z * sm.hcur[b][j];
      sm.hcur[b][j] = hn;
      sm.hb[b][j] = (__bf16)hn;
    }
    __syncthreads();

    // ---- Phase C: attention (each wave owns 2 batch elements) ----
#pragma unroll 1
    for (int e = 0; e < 2; ++e) {
      int slot = wave * 2 + e;
      int bg = bbase + slot;
      // Block-diagonal Q: column h holds q[h*8..h*8+7] at matching K rows, so
      // (key64 @ Bq)[s,h] = sum_d key[h,s,d]*q[h,d] = per-head score.
      v16bf bq0, bq1;
      {
        FragAB f;
#pragma unroll
        for (int j = 0; j < 16; ++j) {
          int K = hi * 16 + j;
          f.v[j] = ((K >> 3) == ln16) ? sm.hb[slot][K] : (__bf16)0.0f;
        }
        bq0 = f.v;
#pragma unroll
        for (int j = 0; j < 16; ++j) {
          int K = 32 + hi * 16 + j;
          f.v[j] = ((K >> 3) == ln16) ? sm.hb[slot][K] : (__bf16)0.0f;
        }
        bq1 = f.v;
      }
      const __bf16* kb = keyb + (size_t)bg * SEQ * HD;
#pragma unroll 2
      for (int mt = 0; mt < SEQ / 16; ++mt) {  // M = seq tile (L2-resident A)
        const __bf16* row = kb + (size_t)(mt * 16 + ln16) * HD;
        v16bf a0 = ld_frag16(row, hi);
        v16bf a1 = ld_frag16(row + 32, hi);
        v8f c = {};
        c = wmma_bf16(a0, bq0, c);
        c = wmma_bf16(a1, bq1, c);
        if (ln16 < NH) {
#pragma unroll
          for (int v = 0; v < 8; ++v)
            sm.scores[slot][ln16][mt * 16 + hi * 8 + v] = (__bf16)c[v];
        }
      }
      // softmax over 288, 4 lanes per head, dword (2x bf16) granularity.
      // Keep unnormalized exp in LDS; scale ctx by rcp(sum) at the end.
      int hg = lane >> 2, lw = lane & 3;
      unsigned int* srow = (unsigned int*)&sm.scores[slot][hg][0];  // 144 dw
      float mx = -1e30f;
      for (int d = lw; d < SEQ / 2; d += 4) {
        unsigned int p = srow[d];
        mx = fmaxf(mx, fmaxf(bflo(p), bfhi(p)));
      }
      mx = fmaxf(mx, __shfl_xor(mx, 1));
      mx = fmaxf(mx, __shfl_xor(mx, 2));
      float sum = 0.f;
      for (int d = lw; d < SEQ / 2; d += 4) {
        unsigned int p = srow[d];
        float e0 = __expf(bflo(p) - mx);
        float e1 = __expf(bfhi(p) - mx);
        sum += e0 + e1;
        srow[d] = f2bfbits(e0) | (f2bfbits(e1) << 16);
      }
      sum += __shfl_xor(sum, 1);
      sum += __shfl_xor(sum, 2);
      // ctx: lane owns k-pair (2*lane, 2*lane+1) -> same head hg = lane>>2,
      // so this lane's own softmax sum is the right normalizer (no shuffle),
      // one att weight serves two MACs, val load is one coalesced dword.
      const unsigned int* vbd = (const unsigned int*)(valb + (size_t)bg * SEQ * HD);
      const unsigned int* arow = (const unsigned int*)&sm.scores[slot][hg][0];
      float acc0 = 0.f, acc1 = 0.f;
#pragma unroll 4
      for (int sd = 0; sd < SEQ / 2; ++sd) {
        unsigned int pa = arow[sd];            // att for s=2sd, 2sd+1
        unsigned int p0 = vbd[(size_t)(2 * sd) * 32 + lane];
        unsigned int p1 = vbd[(size_t)(2 * sd) * 32 + 32 + lane];
        float a0 = bflo(pa), a1 = bfhi(pa);
        acc0 += a0 * bflo(p0);
        acc1 += a0 * bfhi(p0);
        acc0 += a1 * bflo(p1);
        acc1 += a1 * bfhi(p1);
      }
      float rs = fast_rcp(sum);
      ((unsigned int*)&sm.ctxb[slot][0])[lane] =
          f2bfbits(acc0 * rs) | (f2bfbits(acc1 * rs) << 16);
    }
    __syncthreads();

    // ---- Phase D: MLP layer 1 (WMMA) + bias + relu ----
    if (wave < 2) {
      v16bf a0 = ld_frag16(&sm.ctxb[ln16][0], hi);
      v16bf a1 = ld_frag16(&sm.ctxb[ln16][32], hi);
      v8f c = {};
      c = wmma_bf16(a0, bw1a, c);
      c = wmma_bf16(a1, bw1b, c);
      int col = wave * 16 + ln16;
#pragma unroll
      for (int v = 0; v < 8; ++v)
        sm.h1[hi * 8 + v][col] = fmaxf(0.f, c[v] + sm.b1s[col]);
    }
    __syncthreads();

    // ---- Phase E1: scalar head ----
    if (wave == 0 && lane < MT) {
      float acc = b2v;
#pragma unroll
      for (int j = 0; j < 32; ++j) acc += sm.h1[lane][j] * sm.w2s[j];
      sm.outv[lane] = acc;
      out[(size_t)(bbase + lane) * TT + t] = acc;  // (bs, T, 1)
    }
    __syncthreads();

    // ---- Phase E2: next-step input = [x_dec[:,t+1,:28], out] ----
    for (int i = tid; i < MT * 32; i += 256) {
      int b = i >> 5, f = i & 31;
      float xv;
      if (f < DSZ - 1)
        xv = x_dec[(size_t)(bbase + b) * (TT + 1) * DSZ + (size_t)(t + 1) * DSZ + f];
      else if (f == DSZ - 1)
        xv = sm.outv[b];
      else
        xv = 0.f;
      sm.xb[b][f] = (__bf16)xv;
    }
    __syncthreads();
  }
}

extern "C" void kernel_launch(void* const* d_in, const int* in_sizes, int n_in,
                              void* d_out, int out_size, void* d_ws,
                              size_t ws_size, hipStream_t stream) {
  (void)in_sizes; (void)n_in; (void)out_size; (void)ws_size;
  const float* x_dec = (const float*)d_in[0];
  const float* enc   = (const float*)d_in[1];
  const float* hidden = (const float*)d_in[2];
  const float* W_ih = (const float*)d_in[3];
  const float* W_hh = (const float*)d_in[4];
  const float* b_ih = (const float*)d_in[5];
  const float* b_hh = (const float*)d_in[6];
  const float* Wk = (const float*)d_in[7];
  const float* bk = (const float*)d_in[8];
  const float* W1 = (const float*)d_in[9];
  const float* b1 = (const float*)d_in[10];
  const float* W2 = (const float*)d_in[11];
  const float* b2 = (const float*)d_in[12];
  float* outp = (float*)d_out;

  __bf16* valb = (__bf16*)d_ws;                  // 75.5 MB
  __bf16* keyb = valb + (size_t)BSZ * SEQ * HD;  // 75.5 MB (both fit 192MB L2)

  prep_kernel<<<(BSZ * SEQ) / 4, 256, 0, stream>>>(enc, Wk, bk, valb, keyb);
  decoder_step_kernel<<<BSZ / MT, 256, 0, stream>>>(
      x_dec, hidden, W_ih, W_hh, b_ih, b_hh, W1, b1, W2, b2, keyb, valb, outp);
}
